// RBF_76819785056465
// MI455X (gfx1250) — matrix operand order repaired and checked by reading, not compiled, not run on hardware
//
#include <hip/hip_runtime.h>

// Problem constants (match reference setup_inputs)
#define FDIM  1024
#define BROWS 16384
#define CROWS 4096
static constexpr float INV_2S2 = 1.0f / (2.0f * 32.0f * 32.0f);   // 1/(2*sigma^2)

typedef __attribute__((ext_vector_type(16))) __bf16       v16bf;
typedef __attribute__((ext_vector_type(2)))  __bf16       v2bf;
typedef __attribute__((ext_vector_type(8)))  float        v8f;
typedef __attribute__((ext_vector_type(4)))  float        v4f;
typedef __attribute__((ext_vector_type(4)))  unsigned int v4u;

// LDS row stride in u32: 16 data words (32 bf16) + 4 pad words -> spreads banks
#define LDS_STRIDE 20

// Pack two f32 into one u32 of bf16 (RNE) -> v_cvt_pk_bf16_f32
__device__ __forceinline__ unsigned pk2(float lo, float hi) {
#if __has_builtin(__builtin_amdgcn_cvt_pk_bf16_f32)
    auto r = __builtin_amdgcn_cvt_pk_bf16_f32(lo, hi);
    return __builtin_bit_cast(unsigned, r);
#else
    v2bf t;
    t[0] = (__bf16)lo;
    t[1] = (__bf16)hi;
    return __builtin_bit_cast(unsigned, t);
#endif
}

union Frag { v16bf v; v4u u[2]; };

// ---------------------------------------------------------------------------
// Row squared-norm kernel: one wave32 per row, 8 waves per 256-thread block.
// ---------------------------------------------------------------------------
__global__ __launch_bounds__(256) void rbf_row_norms(const float* __restrict__ in,
                                                     float* __restrict__ out,
                                                     int rows) {
    const int wave = (blockIdx.x * blockDim.x + threadIdx.x) >> 5;
    const int lane = threadIdx.x & 31;
    if (wave >= rows) return;
    const float* p = in + (size_t)wave * FDIM;
    float s = 0.0f;
    for (int k = lane * 4; k < FDIM; k += 32 * 4) {
        v4f v = *(const v4f*)(p + k);
        s += v.x * v.x + v.y * v.y + v.z * v.z + v.w * v.w;
    }
    for (int off = 16; off >= 1; off >>= 1) s += __shfl_xor(s, off, 32);
    if (lane == 0) out[wave] = s;
}

// ---------------------------------------------------------------------------
// Fused RBF GEMM:  out[b][c] = exp(-(x2[b] + c2[c] - 2*x.c) / (2 sigma^2))
//
// Block: 256 threads = 8 waves (2x4), tile 128(M) x 256(N), K-step 32.
// Pipeline per K-step:
//   - issue global f32 loads for K-step k+1 (latency hides under WMMAs)
//   - ds_load_b128 bf16 fragments (ISA 7.12.2 lane layout) from LDS buf k&1
//   - 4x4 v_wmma_f32_16x16x32_bf16
//   - cvt_pk f32->bf16 and ds_store_b128 into LDS buf (k+1)&1
//   - one s_barrier (double-buffer => single barrier per iteration is safe)
// ---------------------------------------------------------------------------
__global__ __launch_bounds__(256) void rbf_wmma_kernel(const float* __restrict__ X,
                                                       const float* __restrict__ Cen,
                                                       const float* __restrict__ x2,
                                                       const float* __restrict__ c2,
                                                       float* __restrict__ out) {
    __shared__ unsigned ldsA[2][128 * LDS_STRIDE];   // 128 rows x 32 bf16 (+pad), x2 bufs
    __shared__ unsigned ldsB[2][256 * LDS_STRIDE];   // 256 cols x 32 bf16 (+pad), x2 bufs

    const int t    = threadIdx.x;
    const int lane = t & 31;
    const int wave = t >> 5;
    const int wm   = wave >> 2;          // 0..1
    const int wn   = wave & 3;           // 0..3
    const int l15  = lane & 15;
    const bool hi  = lane >= 16;

    // --- cooperative stage addressing -------------------------------------
    // A: 2 threads per row (t>>1 = row 0..127), each covers 16 K values
    const int arow = (blockIdx.y << 7) + (t >> 1);
    const int akh  = (t & 1) * 16;
    const float* gA = X + (size_t)arow * FDIM + akh;
    const int aoff  = (t >> 1) * LDS_STRIDE + (t & 1) * 8;   // u32 offset in A buf
    // B: 1 thread per column (t = col 0..255), covers all 32 K values
    const float* gB = Cen + (size_t)((blockIdx.x << 8) + t) * FDIM;
    const int boff  = t * LDS_STRIDE;                        // u32 offset in B buf

    v4f ra[4], rb[8];
#pragma unroll
    for (int q = 0; q < 4; ++q) ra[q] = *(const v4f*)(gA + 4 * q);
#pragma unroll
    for (int q = 0; q < 8; ++q) rb[q] = *(const v4f*)(gB + 4 * q);

    // commit K-step 0 into buffer 0
    {
        unsigned* sA = &ldsA[0][aoff];
        unsigned* sB = &ldsB[0][boff];
#pragma unroll
        for (int q = 0; q < 2; ++q) {
            v4u w;
            w[0] = pk2(ra[2 * q][0], ra[2 * q][1]);
            w[1] = pk2(ra[2 * q][2], ra[2 * q][3]);
            w[2] = pk2(ra[2 * q + 1][0], ra[2 * q + 1][1]);
            w[3] = pk2(ra[2 * q + 1][2], ra[2 * q + 1][3]);
            *(v4u*)(sA + 4 * q) = w;
        }
#pragma unroll
        for (int q = 0; q < 4; ++q) {
            v4u w;
            w[0] = pk2(rb[2 * q][0], rb[2 * q][1]);
            w[1] = pk2(rb[2 * q][2], rb[2 * q][3]);
            w[2] = pk2(rb[2 * q + 1][0], rb[2 * q + 1][1]);
            w[3] = pk2(rb[2 * q + 1][2], rb[2 * q + 1][3]);
            *(v4u*)(sB + 4 * q) = w;
        }
    }
    __syncthreads();

    v8f acc[4][4];
#pragma unroll
    for (int i = 0; i < 4; ++i)
#pragma unroll
        for (int j = 0; j < 4; ++j)
            acc[i][j] = {};

    for (int k0 = 0; k0 < FDIM; k0 += 32) {
        const int  buf  = (k0 >> 5) & 1;
        const bool more = (k0 + 32) < FDIM;

        // issue global loads for the next K-step; completion waited only at commit
        if (more) {
            const float* pa = gA + k0 + 32;
            const float* pb = gB + k0 + 32;
#pragma unroll
            for (int q = 0; q < 4; ++q) ra[q] = *(const v4f*)(pa + 4 * q);
#pragma unroll
            for (int q = 0; q < 8; ++q) rb[q] = *(const v4f*)(pb + 4 * q);
            if (k0 + 64 < FDIM) {                       // global_prefetch_b8 for k+2
                __builtin_prefetch(pa + 32, 0, 3);
                __builtin_prefetch(pb + 32, 0, 3);
            }
        }

        // fragment loads from LDS (already in WMMA lane layout)
        const unsigned* Ab = &ldsA[buf][0];
        const unsigned* Bb = &ldsB[buf][0];
        Frag a[4], b[4];
#pragma unroll
        for (int i = 0; i < 4; ++i) {
            const int base = ((wm << 6) + 16 * i + l15) * LDS_STRIDE + (hi ? 4 : 0);
            a[i].u[0] = *(const v4u*)(Ab + base);       // K = aklo .. aklo+7
            a[i].u[1] = *(const v4u*)(Ab + base + 8);   // K = aklo+16 .. aklo+23
        }
#pragma unroll
        for (int j = 0; j < 4; ++j) {
            const int base = ((wn << 6) + 16 * j + l15) * LDS_STRIDE + (hi ? 4 : 0);
            b[j].u[0] = *(const v4u*)(Bb + base);       // K = bklo .. bklo+7
            b[j].u[1] = *(const v4u*)(Bb + base + 8);   // K = bklo+16 .. bklo+23
        }

#pragma unroll
        for (int i = 0; i < 4; ++i)
#pragma unroll
            for (int j = 0; j < 4; ++j)
                acc[i][j] = __builtin_amdgcn_wmma_f32_16x16x32_bf16(
                    false, a[i].v, false, b[j].v,
                    (short)0, acc[i][j], false, false);

        // pack + commit K-step k+1 into the other buffer, then one barrier
        if (more) {
            unsigned* sA = &ldsA[buf ^ 1][aoff];
            unsigned* sB = &ldsB[buf ^ 1][boff];
#pragma unroll
            for (int q = 0; q < 2; ++q) {
                v4u w;
                w[0] = pk2(ra[2 * q][0], ra[2 * q][1]);
                w[1] = pk2(ra[2 * q][2], ra[2 * q][3]);
                w[2] = pk2(ra[2 * q + 1][0], ra[2 * q + 1][1]);
                w[3] = pk2(ra[2 * q + 1][2], ra[2 * q + 1][3]);
                *(v4u*)(sA + 4 * q) = w;
            }
#pragma unroll
            for (int q = 0; q < 4; ++q) {
                v4u w;
                w[0] = pk2(rb[2 * q][0], rb[2 * q][1]);
                w[1] = pk2(rb[2 * q][2], rb[2 * q][3]);
                w[2] = pk2(rb[2 * q + 1][0], rb[2 * q + 1][1]);
                w[3] = pk2(rb[2 * q + 1][2], rb[2 * q + 1][3]);
                *(v4u*)(sB + 4 * q) = w;
            }
            __syncthreads();
        }
    }

    // Epilogue: d2 = x2 + c2 - 2*xc, clamp, exp.  C/D layout: VGPR v holds
    // row m_base+16i + v + 8*(lane>=16), col n_base+16j + (lane&15).
    const int m_base = (blockIdx.y << 7) + (wm << 6);
    const int n_base = (blockIdx.x << 8) + (wn << 6);
    const int mhalf  = hi ? 8 : 0;
#pragma unroll
    for (int i = 0; i < 4; ++i) {
        const int rbase = m_base + 16 * i + mhalf;
        float x2r[8];
#pragma unroll
        for (int v = 0; v < 8; ++v) x2r[v] = x2[rbase + v];
#pragma unroll
        for (int j = 0; j < 4; ++j) {
            const int col = n_base + 16 * j + l15;
            const float c2v = c2[col];
#pragma unroll
            for (int v = 0; v < 8; ++v) {
                float d2 = x2r[v] + c2v - 2.0f * acc[i][j][v];
                d2 = fmaxf(d2, 0.0f);
                out[(size_t)(rbase + v) * CROWS + col] = __expf(-d2 * INV_2S2);
            }
        }
    }
}

// ---------------------------------------------------------------------------
// kernel_launch: norms into d_ws (needs (16384+4096)*4 = 80 KB), then fused
// WMMA GEMM + exp epilogue. All on `stream`, graph-capture safe.
// ---------------------------------------------------------------------------
extern "C" void kernel_launch(void* const* d_in, const int* in_sizes, int n_in,
                              void* d_out, int out_size, void* d_ws, size_t ws_size,
                              hipStream_t stream) {
    const float* X   = (const float*)d_in[0];   // [16384, 1024] f32
    const float* Cen = (const float*)d_in[1];   // [4096, 1024]  f32
    float* out = (float*)d_out;                  // [16384, 4096] f32
    float* x2  = (float*)d_ws;                   // [16384]
    float* c2  = x2 + BROWS;                     // [4096]

    rbf_row_norms<<<BROWS / 8, 256, 0, stream>>>(X,   x2, BROWS);
    rbf_row_norms<<<CROWS / 8, 256, 0, stream>>>(Cen, c2, CROWS);

    dim3 grid(CROWS / 256, BROWS / 128);         // (16, 128)
    rbf_wmma_kernel<<<grid, 256, 0, stream>>>(X, Cen, x2, c2, out);
}
